// Model_86852828660383
// MI455X (gfx1250) — matrix operand order repaired and checked by reading, not compiled, not run on hardware
//
#include <hip/hip_runtime.h>
#include <hip/hip_bf16.h>

// ============================================================================
// MI455X (gfx1250, CDNA5) implementation.
//
//  * wave32 blocks of 128 threads (4 waves)
//  * v_wmma_f32_16x16x32_f16 for every GEMM (f16 in, f32 accumulate)
//  * LSTM recurrence fused into ONE kernel per direction: combined weight
//    [Wih | Whh] (512x256 f16 = 256KB) resident in LDS across all T steps
//    (possible only with CDNA5's 320KB LDS per WGP).
//  * attention Wa (128KB f16) LDS-resident; x tiles staged with
//    GLOBAL_LOAD_ASYNC_TO_LDS_B128 (ASYNCcnt) -- gfx1250 async path.
//  * B fragments batch-loaded into distinct registers per k-step so the
//    8 WMMAs issue back-to-back (partial dscnt waits, no WAR hazard NOPs).
//
// Roofline: ~260 GFLOP total, ~0.5GB unique bytes -> trivially small on
// MI455X (23.3 TB/s); the serial LSTM recurrence dominates, hence fusion.
// ============================================================================

typedef __attribute__((ext_vector_type(16))) _Float16 v16h;
typedef __attribute__((ext_vector_type(8)))  float    v8f;

#define DEVINL __device__ __forceinline__

// ---------------- WMMA fragment loaders (contiguous-k order, same for A and B) ----------------
// A 16x32 f16 (MxK): lane l -> m = l&15 ; halves i: k = i + 16*(l>=16)
DEVINL v16h ld_a16(const _Float16* a, int lda) {
  const int lane = threadIdx.x & 31;
  const int m = lane & 15;
  const int kb = (lane >> 4) << 4;
  v16h r;
#pragma unroll
  for (int i = 0; i < 16; ++i) {
    const int k = i + kb;
    r[i] = a[m * lda + k];
  }
  return r;
}

// B 32x16 f16 (KxN) built from row-major W[n][k] (so B[k][n] = W[n][k]):
// lane l -> n = l&15 ; halves i: k = i + 16*(l>=16)
DEVINL v16h ld_bT16(const _Float16* w, int ldw) {
  const int lane = threadIdx.x & 31;
  const int n = lane & 15;
  const int kb = (lane >> 4) << 4;
  v16h r;
#pragma unroll
  for (int i = 0; i < 16; ++i) r[i] = w[n * ldw + kb + i];
  return r;
}

DEVINL v8f wmma16(v16h a, v16h b, v8f c) {
  return __builtin_amdgcn_wmma_f32_16x16x32_f16(false, a, false, b, (short)0, c,
                                                false, false);
}

DEVINL float sigm(float x) { return 1.f / (1.f + __expf(-x)); }

// monotonic float<->uint encoding for atomicMax on floats (incl. negatives)
DEVINL unsigned fenc(float f) {
  unsigned u = __float_as_uint(f);
  return (u & 0x80000000u) ? ~u : (u | 0x80000000u);
}
DEVINL float fdec(unsigned e) {
  unsigned u = (e & 0x80000000u) ? (e & 0x7FFFFFFFu) : ~e;
  return __uint_as_float(u);
}

// Async global->LDS 16B/lane copy (CDNA5 GLOBAL_LOAD_ASYNC_TO_LDS_B128).
// LDS offset = low 32 bits of generic pointer.
DEVINL void async_g2l_b128(void* lds_dst, const void* gsrc) {
  unsigned ldsa = (unsigned)(unsigned long long)lds_dst;
  unsigned long long ga = (unsigned long long)gsrc;
  asm volatile("global_load_async_to_lds_b128 %0, %1, off"
               :: "v"(ldsa), "v"(ga) : "memory");
}
DEVINL void async_wait0() {
  asm volatile("s_wait_asynccnt 0x0" ::: "memory");
}

// ============================================================================
// Fused BiLSTM direction kernel.
// Block = 128 thr (4 waves), owns 16 batch rows, loops all T steps.
// LDS: sW [512][256] f16 (Wih k<128 | Whh k>=128) = 256KB
//      sA [16][256] f16  (x_t k<128 | h_{t-1} k>=128) = 8KB
//      sB [512] f32 bias = 2KB                      total 272,384 B (<320KB)
// Wave w computes hidden cols j in [32w,32w+32): 8 N-tiles (4 gates x 2).
// Cell state c kept in accumulator-layout registers across all timesteps.
// ============================================================================
__global__ void lstm_dir_kernel(const float* __restrict__ emb,
                                const int* __restrict__ ids,
                                const float* __restrict__ Wih,   // [512][128]
                                const float* __restrict__ Whh,   // [512][128]
                                const float* __restrict__ bias,  // [512]
                                _Float16* __restrict__ out_seq,  // [B][T][256]
                                int T, int reverse, int dir_off) {
  extern __shared__ char smem[];
  _Float16* sW = (_Float16*)smem;                            // 512*256 halves
  _Float16* sA = (_Float16*)(smem + 512 * 256 * 2);          // 16*256 halves
  float* sB    = (float*)(smem + 512 * 256 * 2 + 16 * 256 * 2);

  const int tid = threadIdx.x, wave = tid >> 5, lane = tid & 31;
  const int b0 = blockIdx.x * 16;
  const int jbase = wave * 32;

  for (int idx = tid; idx < 512 * 128; idx += 128) {
    const int n = idx >> 7, k = idx & 127;
    sW[n * 256 + k]       = (_Float16)Wih[idx];
    sW[n * 256 + 128 + k] = (_Float16)Whh[idx];
  }
  for (int idx = tid; idx < 512; idx += 128) sB[idx] = bias[idx];
  for (int idx = tid; idx < 16 * 256; idx += 128) sA[idx] = (_Float16)0.f;
  __syncthreads();

  v8f c0 = {};  // cell state, j tile [jbase, jbase+16)
  v8f c1 = {};  // cell state, j tile [jbase+16, jbase+32)

  for (int t = 0; t < T; ++t) {
    const int te = reverse ? (T - 1 - t) : t;
    // stage x_t (embedding gather, f32 -> f16)
    for (int idx = tid; idx < 16 * 128; idx += 128) {
      const int m = idx >> 7, k = idx & 127;
      const int id = ids[(size_t)(b0 + m) * T + te];
      sA[m * 256 + k] = (_Float16)emb[(size_t)id * 128 + k];
    }
    __syncthreads();

    v8f acc[8] = {};  // [gate*2 + half] : i(0,1) f(2,3) g(4,5) o(6,7)
#pragma unroll 1
    for (int ks = 0; ks < 8; ++ks) {  // K = 256 = 8 * 32
      const v16h a = ld_a16(sA + ks * 32, 256);
      // batch-load all 8 B fragments into distinct registers, then issue
      // the 8 WMMAs back-to-back (partial dscnt waits, no WAR hazards)
      v16h bfr[8];
#pragma unroll
      for (int g = 0; g < 4; ++g) {
#pragma unroll
        for (int h = 0; h < 2; ++h) {
          const int n0 = g * 128 + jbase + h * 16;
          bfr[g * 2 + h] = ld_bT16(sW + (size_t)n0 * 256 + ks * 32, 256);
        }
      }
#pragma unroll
      for (int i = 0; i < 8; ++i) acc[i] = wmma16(a, bfr[i], acc[i]);
    }
    __syncthreads();  // all waves done reading sA before h is overwritten

#pragma unroll
    for (int h = 0; h < 2; ++h) {
      const int nl = jbase + h * 16 + (lane & 15);
      const float bi = sB[nl], bf_ = sB[128 + nl], bg = sB[256 + nl],
                  bo = sB[384 + nl];
      v8f* cp = h ? &c1 : &c0;
#pragma unroll
      for (int r = 0; r < 8; ++r) {
        const float iv = sigm(acc[0 + h][r] + bi);
        const float fv = sigm(acc[2 + h][r] + bf_);
        const float gv = tanhf(acc[4 + h][r] + bg);
        const float ov = sigm(acc[6 + h][r] + bo);
        const float cn = fv * (*cp)[r] + iv * gv;
        (*cp)[r] = cn;
        const float hv = ov * tanhf(cn);
        const int m = r + ((lane >> 4) << 3);
        sA[m * 256 + 128 + nl] = (_Float16)hv;  // h_t for next step
      }
    }
    __syncthreads();

    // cooperative coalesced h -> out_seq (16 rows x 128 halves, b128 stores)
    for (int c = tid; c < 256; c += 128) {  // 256 chunks of 8 halves (16B)
      const int m = c >> 4, o = (c & 15) * 8;
      const uint4 v = *(const uint4*)(sA + m * 256 + 128 + o);
      *(uint4*)(out_seq + ((size_t)(b0 + m) * T + te) * 256 + dir_off + o) = v;
    }
    // no sync needed: next phase writes only the x region of sA; the h region
    // is next written after two more barriers.
  }
}

// ============================================================================
// Fused attention: per batch row b, q = leaky(x@Wa^T + ba, .01),
// s = q@Wp^T + bp, softmax over T, out = sum_t w_t * x_t.
// x tiles staged via async global->LDS b128 (ASYNCcnt).
// ============================================================================
__global__ void attn_kernel(const _Float16* __restrict__ seq,  // [B][T][256]
                            const float* __restrict__ Wa,      // [256][256]
                            const float* __restrict__ ba,      // [256]
                            const float* __restrict__ Wp,      // [256]
                            const float* __restrict__ bp,      // [1]
                            float* __restrict__ out_feat,      // [B][256]
                            int T) {
  extern __shared__ char smem[];
  _Float16* sWa = (_Float16*)smem;                         // 256*256
  _Float16* sX  = (_Float16*)(smem + 256 * 256 * 2);       // 16*256
  float* sBa  = (float*)(smem + 256 * 256 * 2 + 16 * 256 * 2);  // 256
  float* sWp  = sBa + 256;                                 // 256
  float* sS   = sWp + 256;                                 // T
  float* sRed = sS + T;                                    // 128

  const int tid = threadIdx.x, wave = tid >> 5, lane = tid & 31;
  const int b = blockIdx.x;
  const _Float16* x = seq + (size_t)b * T * 256;

  for (int idx = tid; idx < 256 * 256; idx += 128) sWa[idx] = (_Float16)Wa[idx];
  for (int idx = tid; idx < 256; idx += 128) {
    sBa[idx] = ba[idx];
    sWp[idx] = Wp[idx];
  }
  for (int idx = tid; idx < T; idx += 128) sS[idx] = bp[0];
  __syncthreads();

  for (int t0 = 0; t0 < T; t0 += 16) {
    // async-stage 16x256 f16 tile: 8KB = 4 issues of 128 lanes x 16B
#pragma unroll
    for (int it = 0; it < 4; ++it) {
      const int c = tid + it * 128;  // chunk of 8 halves
      async_g2l_b128(sX + c * 8, x + (size_t)t0 * 256 + c * 8);
    }
    async_wait0();
    __syncthreads();

    v8f acc[4] = {};
#pragma unroll 2
    for (int ks = 0; ks < 8; ++ks) {
      const v16h a = ld_a16(sX + ks * 32, 256);
      v16h bfr[4];
#pragma unroll
      for (int nt = 0; nt < 4; ++nt)
        bfr[nt] =
            ld_bT16(sWa + (size_t)(wave * 64 + nt * 16) * 256 + ks * 32, 256);
#pragma unroll
      for (int nt = 0; nt < 4; ++nt) acc[nt] = wmma16(a, bfr[nt], acc[nt]);
    }
#pragma unroll
    for (int nt = 0; nt < 4; ++nt) {
      const int n = wave * 64 + nt * 16 + (lane & 15);
      const float ban = sBa[n], wpn = sWp[n];
#pragma unroll
      for (int r = 0; r < 8; ++r) {
        float q = acc[nt][r] + ban;
        q = q > 0.f ? q : 0.01f * q;  // leaky 0.01
        const int m = r + ((lane >> 4) << 3);
        atomicAdd(&sS[t0 + m], q * wpn);
      }
    }
    __syncthreads();
  }

  // softmax over sS[0..T)
  float lm = -3.4e38f;
  for (int t = tid; t < T; t += 128) lm = fmaxf(lm, sS[t]);
  sRed[tid] = lm;
  __syncthreads();
  for (int s = 64; s > 0; s >>= 1) {
    if (tid < s) sRed[tid] = fmaxf(sRed[tid], sRed[tid + s]);
    __syncthreads();
  }
  const float mx = sRed[0];
  __syncthreads();
  float ls = 0.f;
  for (int t = tid; t < T; t += 128) {
    const float e = __expf(sS[t] - mx);
    sS[t] = e;
    ls += e;
  }
  sRed[tid] = ls;
  __syncthreads();
  for (int s = 64; s > 0; s >>= 1) {
    if (tid < s) sRed[tid] += sRed[tid + s];
    __syncthreads();
  }
  const float denom = sRed[0];
  __syncthreads();

  for (int h = tid; h < 256; h += 128) {
    float a2 = 0.f;
    for (int t = 0; t < T; ++t) a2 += sS[t] * (float)x[(size_t)t * 256 + h];
    out_feat[(size_t)b * 256 + h] = a2 / denom;
  }
}

// ============================================================================
// Generic WMMA GEMM: C[M][256] = act(A[M][K] @ W[256][K]^T + bias), f32 I/O,
// f16 staged in LDS with K zero-padded to Kp (multiple of 32). act: 0 none,
// 1 relu. Block = 128 thr computes 16x256 tile; wave w owns N cols [64w,64w+64).
// ============================================================================
__global__ void gemm256_kernel(const float* __restrict__ A, int M, int K,
                               int Kp, const float* __restrict__ W,
                               const float* __restrict__ bias,
                               float* __restrict__ C, int act) {
  extern __shared__ char smem[];
  _Float16* sW = (_Float16*)smem;                              // [256][Kp]
  _Float16* sA = (_Float16*)(smem + (size_t)256 * Kp * 2);     // [16][Kp]
  float* sBias = (float*)(smem + (size_t)(256 + 16) * Kp * 2);

  const int tid = threadIdx.x, wave = tid >> 5, lane = tid & 31;
  const int m0 = blockIdx.x * 16;

  for (int idx = tid; idx < 256 * Kp; idx += 128) {
    const int n = idx / Kp, k = idx % Kp;
    sW[idx] = (k < K) ? (_Float16)W[(size_t)n * K + k] : (_Float16)0.f;
  }
  for (int idx = tid; idx < 256; idx += 128)
    sBias[idx] = bias ? bias[idx] : 0.f;
  for (int idx = tid; idx < 16 * Kp; idx += 128) {
    const int m = idx / Kp, k = idx % Kp;
    sA[idx] = (m0 + m < M && k < K) ? (_Float16)A[(size_t)(m0 + m) * K + k]
                                    : (_Float16)0.f;
  }
  __syncthreads();

  v8f acc[4] = {};
  const int nks = Kp >> 5;
#pragma unroll 2
  for (int ks = 0; ks < nks; ++ks) {
    const v16h a = ld_a16(sA + ks * 32, Kp);
    v16h bfr[4];
#pragma unroll
    for (int nt = 0; nt < 4; ++nt)
      bfr[nt] =
          ld_bT16(sW + (size_t)(wave * 64 + nt * 16) * Kp + ks * 32, Kp);
#pragma unroll
    for (int nt = 0; nt < 4; ++nt) acc[nt] = wmma16(a, bfr[nt], acc[nt]);
  }
#pragma unroll
  for (int nt = 0; nt < 4; ++nt) {
    const int n = wave * 64 + nt * 16 + (lane & 15);
    const float bn = sBias[n];
#pragma unroll
    for (int r = 0; r < 8; ++r) {
      const int m = m0 + r + ((lane >> 4) << 3);
      if (m < M) {
        float v = acc[nt][r] + bn;
        if (act == 1) v = fmaxf(v, 0.f);
        C[(size_t)m * 256 + n] = v;
      }
    }
  }
}

// ---------------- GAT support kernels ----------------
__global__ void rowdot_kernel(const float* __restrict__ X,
                              const float* __restrict__ v,
                              float* __restrict__ out, int M) {
  __shared__ float red[128];
  const int row = blockIdx.x;
  if (row >= M) return;
  float s = 0.f;
  for (int k = threadIdx.x; k < 256; k += 128)
    s += X[(size_t)row * 256 + k] * v[k];
  red[threadIdx.x] = s;
  __syncthreads();
  for (int st = 64; st > 0; st >>= 1) {
    if (threadIdx.x < st) red[threadIdx.x] += red[threadIdx.x + st];
    __syncthreads();
  }
  if (threadIdx.x == 0) out[row] = red[0];
}

__global__ void seg_init_kernel(unsigned* __restrict__ menc,
                                float* __restrict__ denom, int n) {
  const int i = blockIdx.x * blockDim.x + threadIdx.x;
  if (i < n) {
    menc[i] = 0x007FFFFFu;  // enc(-inf)
    denom[i] = 0.f;
  }
}

__global__ void gat_e1_kernel(const int* __restrict__ edge, int E,
                              const float* __restrict__ a_s,
                              const float* __restrict__ a_d,
                              float* __restrict__ ebuf,
                              unsigned* __restrict__ menc) {
  const int i = blockIdx.x * blockDim.x + threadIdx.x;
  if (i >= E) return;
  const int s = edge[i], d = edge[E + i];
  float e = a_s[s] + a_d[d];
  e = e > 0.f ? e : 0.2f * e;  // leaky 0.2
  ebuf[i] = e;
  atomicMax(&menc[d], fenc(e));
}

__global__ void gat_e2_kernel(const int* __restrict__ edge, int E,
                              const float* __restrict__ ebuf,
                              const unsigned* __restrict__ menc,
                              float* __restrict__ eebuf,
                              float* __restrict__ denom) {
  const int i = blockIdx.x * blockDim.x + threadIdx.x;
  if (i >= E) return;
  const int d = edge[E + i];
  float m = fdec(menc[d]);
  if (!(m > -1e30f && m < 1e30f)) m = 0.f;  // isfinite guard
  const float ee = __expf(ebuf[i] - m);
  eebuf[i] = ee;
  atomicAdd(&denom[d], ee);
}

__global__ void gat_scatter_kernel(const int* __restrict__ edge, int E,
                                   const float* __restrict__ eebuf,
                                   const float* __restrict__ denom,
                                   const float* __restrict__ hs,
                                   float* __restrict__ out) {
  const long long gid = (long long)blockIdx.x * blockDim.x + threadIdx.x;
  if (gid >= (long long)E * 256) return;
  const int i = (int)(gid >> 8);
  const int h = (int)(gid & 255);
  const int s = edge[i], d = edge[E + i];
  const float alpha = eebuf[i] / denom[d];
  atomicAdd(&out[(size_t)d * 256 + h], alpha * hs[(size_t)s * 256 + h]);
}

__global__ void combine_kernel(float* __restrict__ buf,
                               const float* __restrict__ b1,
                               const float* __restrict__ b2, int M, int relu) {
  const long long gid = (long long)blockIdx.x * blockDim.x + threadIdx.x;
  if (gid >= (long long)M * 256) return;
  const int n = (int)(gid & 255);
  float v = buf[gid] + (b1 ? b1[n] : 0.f) + (b2 ? b2[n] : 0.f);
  if (relu) v = fmaxf(v, 0.f);
  buf[gid] = v;
}

__global__ void fusion_kernel(const float* __restrict__ doc_feat,
                              const float* __restrict__ g_doc,
                              const float* __restrict__ bert_feat,
                              const float* __restrict__ Wf,  // [2][768]
                              const float* __restrict__ bf,  // [2]
                              float* __restrict__ out) {     // [B][2]
  __shared__ float red[256];
  const int b = blockIdx.x, tid = threadIdx.x;
  float s0 = 0.f, s1 = 0.f;
  for (int k = tid; k < 768; k += 128) {
    const float v = (k < 256)   ? doc_feat[(size_t)b * 256 + k]
                    : (k < 512) ? g_doc[(size_t)b * 256 + (k - 256)]
                                : bert_feat[(size_t)b * 256 + (k - 512)];
    s0 += v * Wf[k];
    s1 += v * Wf[768 + k];
  }
  red[tid] = s0;
  red[128 + tid] = s1;
  __syncthreads();
  for (int st = 64; st > 0; st >>= 1) {
    if (tid < st) {
      red[tid] += red[tid + st];
      red[128 + tid] += red[128 + tid + st];
    }
    __syncthreads();
  }
  if (tid == 0) {
    out[(size_t)b * 2 + 0] = red[0] + bf[0];
    out[(size_t)b * 2 + 1] = red[128] + bf[1];
  }
}

// ============================================================================
// Host orchestration
// ============================================================================
extern "C" void kernel_launch(void* const* d_in, const int* in_sizes, int n_in,
                              void* d_out, int out_size, void* d_ws,
                              size_t ws_size, hipStream_t stream) {
  (void)in_sizes; (void)n_in; (void)out_size; (void)ws_size;

  const int B = 1024, T1 = 256, T2 = 128;
  const int NW = 5000, NR = 10000, FW = 300, FR = 300;
  const int EWD = 51200, ERD = 20480;

  // --- input leaf indices (params pytree flattened in dict-insertion order) ---
  // 0 emb1, 1 emb2
  // lstm1: 2 Wih_f 3 Whh_f 4 b_f 5 Wih_b 6 Whh_b 7 b_b ; lstm2: 8..13
  // attn1: 14 Wa 15 ba 16 Wp 17 bp ; attn2: 18..21
  // 22 mlp_word.W 23 .b ; 24 mlp_res.W 25 .b
  // gat p = {Ws,Wd,as,ad,b}: g1_wd 26..30, g1_rd 31..35, g1_dw 36..40,
  //                          g1_dr 41..45, g2_wd 46..50, g2_rd 51..55
  // 56/57 lin1_doc, 58/59 lin1_word, 60/61 lin1_resp, 62/63 lin2_doc
  // 64 fusion.W 65 fusion.b
  // 66 word_x 67 resp_x 68 ids 69 bert_ids 70 length
  // 71 edge_wd 72 edge_dw 73 edge_rd 74 edge_dr 75 abl
  auto F = [&](int i) { return (const float*)d_in[i]; };
  auto I = [&](int i) { return (const int*)d_in[i]; };

  // --- workspace carve (~260MB) ---
  char* ws = (char*)d_ws;
  size_t off = 0;
  auto alloc = [&](size_t bytes) {
    void* p = ws + off;
    off += bytes;
    off = (off + 255) & ~(size_t)255;
    return p;
  };
  _Float16* doc_seq  = (_Float16*)alloc((size_t)B * T1 * 256 * 2);
  _Float16* bert_seq = (_Float16*)alloc((size_t)B * T2 * 256 * 2);
  float* doc_feat  = (float*)alloc((size_t)B * 256 * 4);
  float* bert_feat = (float*)alloc((size_t)B * 256 * 4);
  float* word_feat = (float*)alloc((size_t)NW * 256 * 4);
  float* resp_feat = (float*)alloc((size_t)NR * 256 * 4);
  float* h_doc  = (float*)alloc((size_t)B * 256 * 4);
  float* h_word = (float*)alloc((size_t)NW * 256 * 4);
  float* h_resp = (float*)alloc((size_t)NR * 256 * 4);
  float* g_doc  = (float*)alloc((size_t)B * 256 * 4);
  float* hs_buf = (float*)alloc((size_t)NR * 256 * 4);
  float* hd_buf = (float*)alloc((size_t)NR * 256 * 4);
  float* a_s = (float*)alloc((size_t)NR * 4);
  float* a_d = (float*)alloc((size_t)NR * 4);
  unsigned* menc = (unsigned*)alloc((size_t)NR * 4);
  float* denom = (float*)alloc((size_t)NR * 4);
  float* ebuf  = (float*)alloc((size_t)EWD * 4);
  float* eebuf = (float*)alloc((size_t)EWD * 4);

  // --- BiLSTMs (4 fused recurrence kernels; 256KB weights in LDS each) ---
  const size_t LSTM_SMEM = 512 * 256 * 2 + 16 * 256 * 2 + 512 * 4;
  lstm_dir_kernel<<<B / 16, 128, LSTM_SMEM, stream>>>(
      F(0), I(68), F(2), F(3), F(4), doc_seq, T1, 0, 0);
  lstm_dir_kernel<<<B / 16, 128, LSTM_SMEM, stream>>>(
      F(0), I(68), F(5), F(6), F(7), doc_seq, T1, 1, 128);
  lstm_dir_kernel<<<B / 16, 128, LSTM_SMEM, stream>>>(
      F(1), I(69), F(8), F(9), F(10), bert_seq, T2, 0, 0);
  lstm_dir_kernel<<<B / 16, 128, LSTM_SMEM, stream>>>(
      F(1), I(69), F(11), F(12), F(13), bert_seq, T2, 1, 128);

  // --- fused attention ---
  auto attn_smem = [&](int T) {
    return (size_t)256 * 256 * 2 + 16 * 256 * 2 + (256 + 256 + T + 128) * 4;
  };
  attn_kernel<<<B, 128, attn_smem(T1), stream>>>(doc_seq, F(14), F(15), F(16),
                                                 F(17), doc_feat, T1);
  attn_kernel<<<B, 128, attn_smem(T2), stream>>>(bert_seq, F(18), F(19), F(20),
                                                 F(21), bert_feat, T2);

  // --- feature MLPs (K=300 padded to 320) ---
  auto gsmem = [&](int Kp) { return (size_t)(256 + 16) * Kp * 2 + 256 * 4; };
  gemm256_kernel<<<(NW + 15) / 16, 128, gsmem(320), stream>>>(
      F(66), NW, FW, 320, F(22), F(23), word_feat, 1);
  gemm256_kernel<<<(NR + 15) / 16, 128, gsmem(320), stream>>>(
      F(67), NR, FR, 320, F(24), F(25), resp_feat, 1);

  // --- GAT conv helper (accumulates segment-softmax output into out_accum) ---
  auto run_gat = [&](const float* xs, int Ms, const float* xd, int Md,
                     const int* edge, int E, const float* Wsrc,
                     const float* Wdst, const float* avs, const float* avd,
                     float* out_accum) {
    gemm256_kernel<<<(Ms + 15) / 16, 128, gsmem(256), stream>>>(
        xs, Ms, 256, 256, Wsrc, nullptr, hs_buf, 0);
    gemm256_kernel<<<(Md + 15) / 16, 128, gsmem(256), stream>>>(
        xd, Md, 256, 256, Wdst, nullptr, hd_buf, 0);
    rowdot_kernel<<<Ms, 128, 0, stream>>>(hs_buf, avs, a_s, Ms);
    rowdot_kernel<<<Md, 128, 0, stream>>>(hd_buf, avd, a_d, Md);
    seg_init_kernel<<<(Md + 255) / 256, 256, 0, stream>>>(menc, denom, Md);
    gat_e1_kernel<<<(E + 255) / 256, 256, 0, stream>>>(edge, E, a_s, a_d, ebuf,
                                                       menc);
    gat_e2_kernel<<<(E + 255) / 256, 256, 0, stream>>>(edge, E, ebuf, menc,
                                                       eebuf, denom);
    const long long tot = (long long)E * 256;
    gat_scatter_kernel<<<(unsigned)((tot + 255) / 256), 256, 0, stream>>>(
        edge, E, eebuf, denom, hs_buf, out_accum);
  };

  // --- layer 1 ---
  gemm256_kernel<<<(B + 15) / 16, 128, gsmem(256), stream>>>(
      doc_feat, B, 256, 256, F(56), F(57), h_doc, 0);  // lin1_doc
  run_gat(word_feat, NW, doc_feat, B, I(71), EWD, F(26), F(27), F(28), F(29),
          h_doc);  // g1_wd
  run_gat(resp_feat, NR, doc_feat, B, I(73), ERD, F(31), F(32), F(33), F(34),
          h_doc);  // g1_rd
  combine_kernel<<<(B * 256 + 255) / 256, 256, 0, stream>>>(h_doc, F(30),
                                                            F(35), B, 1);

  gemm256_kernel<<<(NW + 15) / 16, 128, gsmem(256), stream>>>(
      word_feat, NW, 256, 256, F(58), F(59), h_word, 0);  // lin1_word
  run_gat(doc_feat, B, word_feat, NW, I(72), EWD, F(36), F(37), F(38), F(39),
          h_word);  // g1_dw
  combine_kernel<<<(NW * 256 + 255) / 256, 256, 0, stream>>>(h_word, F(40),
                                                             nullptr, NW, 1);

  gemm256_kernel<<<(NR + 15) / 16, 128, gsmem(256), stream>>>(
      resp_feat, NR, 256, 256, F(60), F(61), h_resp, 0);  // lin1_resp
  run_gat(doc_feat, B, resp_feat, NR, I(74), ERD, F(41), F(42), F(43), F(44),
          h_resp);  // g1_dr
  combine_kernel<<<(NR * 256 + 255) / 256, 256, 0, stream>>>(h_resp, F(45),
                                                             nullptr, NR, 1);

  // --- layer 2 ---
  gemm256_kernel<<<(B + 15) / 16, 128, gsmem(256), stream>>>(
      h_doc, B, 256, 256, F(62), F(63), g_doc, 0);  // lin2_doc
  run_gat(h_word, NW, h_doc, B, I(71), EWD, F(46), F(47), F(48), F(49),
          g_doc);  // g2_wd
  run_gat(h_resp, NR, h_doc, B, I(73), ERD, F(51), F(52), F(53), F(54),
          g_doc);  // g2_rd
  combine_kernel<<<(B * 256 + 255) / 256, 256, 0, stream>>>(g_doc, F(50),
                                                            F(55), B, 0);

  // --- fusion head ---
  fusion_kernel<<<B, 128, 0, stream>>>(doc_feat, g_doc, bert_feat, F(64),
                                       F(65), (float*)d_out);
}
